// SageAggregator_23055384445490
// MI455X (gfx1250) — compile-verified, matrix operand order checked by reading, not run on hardware
//
#include <hip/hip_runtime.h>
#include <stdint.h>

#define NNODES 16384
#define KN 25

typedef __attribute__((ext_vector_type(16))) __bf16 v16bf;
typedef __attribute__((ext_vector_type(8)))  float  v8f;

union FragBF { v16bf v; uint32_t u[8]; };
union FW { float f; uint32_t u; };

__device__ __forceinline__ uint16_t f2bf(float x) {
  FW w; w.f = x;
  uint32_t r = w.u + 0x7FFFu + ((w.u >> 16) & 1u);   // round-to-nearest-even
  return (uint16_t)(r >> 16);
}
__device__ __forceinline__ uint32_t pack2bf(float lo, float hi) {
  return (uint32_t)f2bf(lo) | ((uint32_t)f2bf(hi) << 16);
}
__device__ __forceinline__ float bf2f(uint16_t h) {
  FW w; w.u = ((uint32_t)h) << 16;
  return w.f;
}
// fast sigmoid: v_exp_f32 + v_rcp_f32, no IEEE divide sequence
__device__ __forceinline__ float sigmoidf_(float x) {
  return __builtin_amdgcn_rcpf(1.0f + __expf(-x));
}
__device__ __forceinline__ float waveRed(float v) {
  #pragma unroll
  for (int off = 16; off > 0; off >>= 1) v += __shfl_xor(v, off, 32);
  return v;
}
// K index held by VGPR j (pair base), per 16-bit WMMA operand layout
__device__ __forceinline__ int kfrag(int j, int half) {
  return (j < 4) ? (8 * half + 2 * j) : (16 + 8 * half + 2 * (j - 4));
}

// ---------------------------------------------------------------------------
// Pass 0: convert weights to bf16, column-major packed (k contiguous per col)
// ---------------------------------------------------------------------------
__global__ void prep_weights(const float* __restrict__ Wlink,   // [64][256]
                             const float* __restrict__ Wself,   // [256][256]
                             const float* __restrict__ Wneigh,  // [256][256]
                             uint16_t* __restrict__ Wlt,        // [256][64]
                             uint16_t* __restrict__ Wst,        // [256][256]
                             uint16_t* __restrict__ Wnt) {      // [256][256]
  int tid = blockIdx.x * blockDim.x + threadIdx.x;   // 0..65535
  int k = tid >> 8, c = tid & 255;
  Wst[c * 256 + k] = f2bf(Wself[tid]);
  Wnt[c * 256 + k] = f2bf(Wneigh[tid]);
  if (tid < 64 * 256) {                              // k < 64 here
    Wlt[c * 64 + k] = f2bf(Wlink[tid]);
  }
}

// ---------------------------------------------------------------------------
// Pass 1: per node — gates + (linkK x 64)@(64 x 256) bf16 WMMA GEMM fused with
// sigmoid * neigh * (gate/p/K), mean over k -> expected[N,256]
// ---------------------------------------------------------------------------
__global__ void __launch_bounds__(256)
sage_phaseA(const float* __restrict__ selfv,
            const float* __restrict__ neigh,
            const float* __restrict__ link,
            const float* __restrict__ probs,
            const float* __restrict__ gself,
            const float* __restrict__ gneigh,
            const float* __restrict__ glink,
            const uint32_t* __restrict__ WltU,   // bf16 colmajor [256][64] packed
            float* __restrict__ expected) {
  __shared__ uint16_t sWt[256 * 66];   // W_link^T, padded stride 66 halves
  __shared__ uint16_t sA[32 * 66];     // link rows (25 real + 7 zero), padded
  __shared__ uint16_t sNeigh[KN * 256];
  __shared__ float    sExpP[4 * 256];  // exclusive partials: (mtile*2+half) x col
  __shared__ float    sTSp[8];
  __shared__ float    sw[32];

  const int n    = blockIdx.x;
  const int tid  = threadIdx.x;
  const int wv   = tid >> 5;
  const int lane = tid & 31;
  const int half = lane >> 4;
  const int l16  = lane & 15;

  // ---- stage LDS (packed bf16-pair stores) ----
  uint32_t* sWtU = (uint32_t*)sWt;
  uint32_t* sAU  = (uint32_t*)sA;
  uint32_t* sNU  = (uint32_t*)sNeigh;

  for (int i = 0; i < 32; ++i) {
    int uidx = tid + 256 * i;               // 8192 dwords total
    int nn = uidx >> 5, kp = uidx & 31;
    sWtU[nn * 33 + kp] = WltU[uidx];
  }
  const float2* ng = (const float2*)(neigh + (size_t)n * (KN * 256));
  for (int p = tid; p < KN * 128; p += 256)             // 3200 pairs
    sNU[p] = pack2bf(ng[p].x, ng[p].y);
  const float2* lk = (const float2*)(link + (size_t)n * (KN * 64));
  for (int p = tid; p < KN * 32; p += 256) {            // 800 pairs
    float2 v = lk[p];
    sAU[(p >> 5) * 33 + (p & 31)] = pack2bf(v.x, v.y);  // stride 33 dwords
  }
  for (int i2 = KN * 33 + tid; i2 < 32 * 33; i2 += 256) sAU[i2] = 0;  // pad rows
  if (tid >= KN && tid < 32) sw[tid] = 0.f;             // pad gate weights

  // self-gate partial: deterministic per-wave slots, fixed-order combine
  float p = selfv[(size_t)n * 256 + tid] * gself[tid];
  p = waveRed(p);
  if (lane == 0) sTSp[wv] = p;
  __syncthreads();

  float tself = 0.f;
  #pragma unroll
  for (int i = 0; i < 8; ++i) tself += sTSp[i];

  // gate per neighbor (k strided over waves)
  for (int k = wv; k < KN; k += 8) {
    float acc = 0.f;
    #pragma unroll
    for (int j = 0; j < 8; ++j) {
      int d = lane + 32 * j;
      acc += bf2f(sNeigh[k * 256 + d]) * gneigh[d];
    }
    #pragma unroll
    for (int j = 0; j < 2; ++j) {
      int d = lane + 32 * j;
      acc += bf2f(sA[k * 66 + d]) * glink[d];
    }
    acc = waveRed(acc);
    if (lane == 0) {
      float g = sigmoidf_(tself + acc);
      sw[k] = g * __builtin_amdgcn_rcpf(probs[(size_t)n * KN + k]) *
              (1.0f / (float)KN);
    }
  }
  __syncthreads();

  // 32 output tiles (mtile in {0,1} over padded link rows, ntile 0..15), 4/wave
  #pragma unroll
  for (int t = 0; t < 4; ++t) {
    int tileIdx = wv * 4 + t;
    int mtile = tileIdx >> 4;
    int ntile = tileIdx & 15;
    int col = ntile * 16 + l16;
    v8f acc = {0.f, 0.f, 0.f, 0.f, 0.f, 0.f, 0.f, 0.f};
    #pragma unroll
    for (int ks = 0; ks < 64; ks += 32) {
      FragBF a, b;
      #pragma unroll
      for (int j = 0; j < 8; ++j) {
        int k = ks + kfrag(j, half);
        a.u[j] = *(const uint32_t*)&sA[(mtile * 16 + l16) * 66 + k];
        b.u[j] = *(const uint32_t*)&sWt[col * 66 + k];
      }
      acc = __builtin_amdgcn_wmma_f32_16x16x32_bf16(false, a.v, false, b.v,
                                                    (short)0, acc, false, false);
    }
    // fused, branchless: pad rows get sw==0; clamp row for the LDS read
    float psum = 0.f;
    #pragma unroll
    for (int i = 0; i < 8; ++i) {
      int krow = mtile * 16 + half * 8 + i;   // C layout: M = i + 8*half
      int krc  = krow < KN ? krow : (KN - 1);
      float s = sigmoidf_(acc[i]);
      psum += s * bf2f(sNeigh[krc * 256 + col]) * sw[krow];
    }
    sExpP[(mtile * 2 + half) * 256 + col] = psum;   // exclusive slot, no atomics
  }
  __syncthreads();

  expected[(size_t)n * 256 + tid] =
      (sExpP[tid] + sExpP[256 + tid]) + (sExpP[512 + tid] + sExpP[768 + tid]);
}

// ---------------------------------------------------------------------------
// Pass 2: out = relu([self@W_self, expected@W_neigh]); 16 rows per block,
// waves 0-3 self half, waves 4-7 neigh half; 8 bf16 WMMAs per tile (K=256)
// ---------------------------------------------------------------------------
__global__ void __launch_bounds__(256)
sage_phaseB(const float* __restrict__ selfv,
            const float* __restrict__ expected,
            const uint32_t* __restrict__ WstU,   // bf16 colmajor [256][256] packed
            const uint32_t* __restrict__ WntU,
            float* __restrict__ out) {
  __shared__ uint16_t sAs[16 * 258];
  __shared__ uint16_t sAe[16 * 258];
  const int rb   = blockIdx.x;
  const int tid  = threadIdx.x;
  const int wv   = tid >> 5;
  const int lane = tid & 31;
  const int half = lane >> 4;
  const int l16  = lane & 15;

  uint32_t* sAsU = (uint32_t*)sAs;
  uint32_t* sAeU = (uint32_t*)sAe;
  const float2* sv = (const float2*)(selfv    + (size_t)rb * 16 * 256);
  const float2* ev = (const float2*)(expected + (size_t)rb * 16 * 256);
  for (int pp = tid; pp < 2048; pp += 256) {            // 16 rows x 128 pairs
    int r = pp >> 7, kp = pp & 127;
    float2 a = sv[pp];
    float2 b = ev[pp];
    sAsU[r * 129 + kp] = pack2bf(a.x, a.y);             // stride 258 halves
    sAeU[r * 129 + kp] = pack2bf(b.x, b.y);
  }
  __syncthreads();

  #pragma unroll
  for (int t = 0; t < 4; ++t) {
    int nt = wv * 4 + t;                   // 0..31, wave-uniform
    bool shalf = (nt < 16);
    const uint16_t* Am = shalf ? sAs : sAe;
    const uint32_t* Bg = shalf ? WstU : WntU;
    int ntl  = shalf ? nt : (nt - 16);
    int col  = nt * 16 + l16;              // output column 0..511
    int bcol = ntl * 16 + l16;             // weight column 0..255
    v8f acc = {0.f, 0.f, 0.f, 0.f, 0.f, 0.f, 0.f, 0.f};
    #pragma unroll
    for (int ks = 0; ks < 256; ks += 32) {
      FragBF a, b;
      #pragma unroll
      for (int j = 0; j < 8; ++j) {
        int k = ks + kfrag(j, half);
        a.u[j] = *(const uint32_t*)&Am[l16 * 258 + k];
        b.u[j] = Bg[bcol * 128 + (k >> 1)];   // L2-resident weights
      }
      acc = __builtin_amdgcn_wmma_f32_16x16x32_bf16(false, a.v, false, b.v,
                                                    (short)0, acc, false, false);
    }
    #pragma unroll
    for (int i = 0; i < 8; ++i) {
      int r = half * 8 + i;
      float v = acc[i];
      out[(size_t)(rb * 16 + r) * 512 + col] = v > 0.f ? v : 0.f;
    }
  }
}

// ---------------------------------------------------------------------------
extern "C" void kernel_launch(void* const* d_in, const int* in_sizes, int n_in,
                              void* d_out, int out_size, void* d_ws, size_t ws_size,
                              hipStream_t stream) {
  (void)in_sizes; (void)n_in; (void)out_size; (void)ws_size;
  const float* selfv  = (const float*)d_in[0];
  const float* neigh  = (const float*)d_in[1];
  const float* link   = (const float*)d_in[2];
  const float* probs  = (const float*)d_in[3];
  const float* gself  = (const float*)d_in[4];
  const float* gneigh = (const float*)d_in[5];
  const float* glink  = (const float*)d_in[6];
  const float* Wlink  = (const float*)d_in[7];
  const float* Wself  = (const float*)d_in[8];
  const float* Wneigh = (const float*)d_in[9];
  float* out = (float*)d_out;

  char* ws = (char*)d_ws;
  float*    expected = (float*)ws;                                   // 16 MB
  uint16_t* Wlt = (uint16_t*)(ws + (size_t)NNODES * 256 * 4);        // 32 KB
  uint16_t* Wst = (uint16_t*)((char*)Wlt + 256 * 64 * 2);            // 128 KB
  uint16_t* Wnt = (uint16_t*)((char*)Wst + 256 * 256 * 2);           // 128 KB

  prep_weights<<<256, 256, 0, stream>>>(Wlink, Wself, Wneigh, Wlt, Wst, Wnt);
  sage_phaseA<<<NNODES, 256, 0, stream>>>(selfv, neigh, link, probs,
                                          gself, gneigh, glink,
                                          (const uint32_t*)Wlt, expected);
  sage_phaseB<<<NNODES / 16, 256, 0, stream>>>(selfv, expected,
                                               (const uint32_t*)Wst,
                                               (const uint32_t*)Wnt, out);
}
